// CustomAttention_38543036514924
// MI455X (gfx1250) — matrix operand order repaired and checked by reading, compiled, not run on hardware
//
#include <hip/hip_runtime.h>

typedef unsigned short u16;
typedef unsigned int   u32;
typedef unsigned long long u64;
typedef __bf16 bf16;
typedef __attribute__((ext_vector_type(16))) bf16  v16bf;
typedef __attribute__((ext_vector_type(8)))  float v8f;
typedef unsigned int u32x4 __attribute__((ext_vector_type(4)));
typedef int          i32x4 __attribute__((ext_vector_type(4)));
typedef int          i32x8 __attribute__((ext_vector_type(8)));

#define BB   64
#define NN   197
#define CC   768
#define HH   12
#define DD   64
#define NG   4
#define GSZ  49
#define MTOT (BB*NN)     // 12608
#define SCALEF 0.125f

union FragU { v16bf v; uint4 q[2]; };

__device__ __forceinline__ float bf2f(u16 u) {
  union { u32 i; float f; } x; x.i = ((u32)u) << 16; return x.f;
}
__device__ __forceinline__ u16 f2bf(float f) {
  union { float f; u32 i; } x; x.f = f;
  u32 i = x.i + 0x7FFFu + ((x.i >> 16) & 1u);
  return (u16)(i >> 16);
}
__device__ __forceinline__ v8f wmma_bf16(v16bf a, v16bf b, v8f c) {
  // D = A(16x32 bf16) * B(32x16 bf16) + C(16x16 f32)
  return __builtin_amdgcn_wmma_f32_16x16x32_bf16(false, a, false, b, (short)0, c, false, false);
}
// LDS-relative byte offset of a __shared__ object (generic -> AS3 -> int)
__device__ __forceinline__ u32 lds_off(const void* p) {
  typedef __attribute__((address_space(3))) const void* lds_ptr_t;
  return (u32)(size_t)(lds_ptr_t)p;
}
// async global->LDS 16B copy, tracked by ASYNCcnt
__device__ __forceinline__ void async_copy_b128(u32 lds_byte_off, const void* gsrc) {
  asm volatile("global_load_async_to_lds_b128 %0, %1, off"
               :: "v"(lds_byte_off), "v"((u64)(size_t)gsrc)
               : "memory");
}
__device__ __forceinline__ void wait_asynccnt0() {
  asm volatile("s_wait_asynccnt 0x0" ::: "memory");
}

// ---------------------------------------------------------------- converts
__global__ void __launch_bounds__(256) cvt_f32_bf16(const float* __restrict__ in,
                                                    u16* __restrict__ out, int n) {
  int i = blockIdx.x * 256 + threadIdx.x;
  if (i < n) out[i] = f2bf(in[i]);
}

// ---------------------------------------------------------------- GEMM core
// C[M x Ndim] = A[M x 768](bf16) * W[768 x Ndim](bf16) + bias
// mode 0: scatter into q/k/v  [B,H,N,D] bf16      (Ndim = 2304)
// mode 1: store f32 row-major into Out             (Ndim = 768)
__global__ void __launch_bounds__(256) gemm_bf16(
    const u16* __restrict__ A, const u16* __restrict__ W, const float* __restrict__ bias,
    int Ndim, int mode,
    u16* __restrict__ Qo, u16* __restrict__ Ko, u16* __restrict__ Vo,
    float* __restrict__ Out)
{
  __shared__ __align__(16) u16 As[128 * 40];   // [row][k], stride 40 (80B, 16B-aligned)
  __shared__ __align__(16) u16 Bs[64 * 40];    // transposed: [n][k], stride 40

  const int tid  = threadIdx.x;
  const int lane = tid & 31, wid = tid >> 5;
  const int wm = wid & 3, wn = wid >> 2;       // 4x2 waves -> 128x64 tile
  const int tileM = blockIdx.y * 128, tileN = blockIdx.x * 64;

  v8f acc[2][2];
  #pragma unroll
  for (int i = 0; i < 2; ++i)
    #pragma unroll
    for (int j = 0; j < 2; ++j) acc[i][j] = (v8f){0,0,0,0,0,0,0,0};

  for (int kt = 0; kt < 24; ++kt) {            // K = 768 = 24 * 32
    // ---- A tile 128x32: async global->LDS, 16B per lane, 2 passes
    {
      int cg = tid & 3;
      #pragma unroll
      for (int p = 0; p < 2; ++p) {
        int row = (tid >> 2) + p * 64;
        int gr = tileM + row; if (gr > MTOT - 1) gr = MTOT - 1;
        const u16* src = A + (size_t)gr * 768 + kt * 32 + cg * 8;
        async_copy_b128(lds_off(As + row * 40 + cg * 8), src);
        if (kt + 1 < 24) __builtin_prefetch(src + 32, 0, 1); // global_prefetch_b8
      }
    }
    // ---- B tile 32x64 -> LDS transposed [n][k]
    #pragma unroll
    for (int p = 0; p < 8; ++p) {
      int idx = tid + p * 256;
      int k = idx >> 6, n = idx & 63;
      Bs[n * 40 + k] = W[(size_t)(kt * 32 + k) * Ndim + tileN + n];
    }
    wait_asynccnt0();
    __syncthreads();

    FragU af[2], bw[2];
    #pragma unroll
    for (int sm = 0; sm < 2; ++sm) {
      int r  = wm * 32 + sm * 16 + (lane & 15);
      int k0 = (lane & 16) ? 8 : 0;            // A-frag: runs K[k0..k0+7], K[k0+16..k0+23]
      af[sm].q[0] = *(const uint4*)(As + r * 40 + k0);
      af[sm].q[1] = *(const uint4*)(As + r * 40 + k0 + 16);
    }
    #pragma unroll
    for (int sn = 0; sn < 2; ++sn) {
      int n  = wn * 32 + sn * 16 + (lane & 15);
      int k0 = (lane & 16) ? 8 : 0;            // B-frag: runs K[k0..k0+7], K[k0+16..k0+23]
      bw[sn].q[0] = *(const uint4*)(Bs + n * 40 + k0);
      bw[sn].q[1] = *(const uint4*)(Bs + n * 40 + k0 + 16);
    }
    #pragma unroll
    for (int sm = 0; sm < 2; ++sm)
      #pragma unroll
      for (int sn = 0; sn < 2; ++sn)
        acc[sm][sn] = wmma_bf16(af[sm].v, bw[sn].v, acc[sm][sn]);
    __syncthreads();
  }

  // ---- epilogue
  #pragma unroll
  for (int sm = 0; sm < 2; ++sm)
    #pragma unroll
    for (int sn = 0; sn < 2; ++sn) {
      int col = tileN + wn * 32 + sn * 16 + (lane & 15);
      float bv = bias[col];
      #pragma unroll
      for (int p = 0; p < 8; ++p) {
        int grow = tileM + wm * 32 + sm * 16 + p + ((lane & 16) ? 8 : 0);
        if (grow >= MTOT) continue;
        float val = acc[sm][sn][p] + bv;
        if (mode == 0) {
          int s = col / 768, rem = col % 768;
          int h = rem >> 6, d = rem & 63;
          int b = grow / 197, n = grow % 197;
          size_t dst = ((size_t)(b * HH + h) * NN + n) * DD + d;
          u16 hv = f2bf(val);
          if (s == 0) Qo[dst] = hv; else if (s == 1) Ko[dst] = hv; else Vo[dst] = hv;
        } else {
          Out[(size_t)grow * 768 + col] = val;
        }
      }
    }
}

// ---------------------------------------------------------------- attention
// One workgroup per (b,h). Keys/queries padded to 224.
__global__ void __launch_bounds__(256) attn_bf16(
    const u16* __restrict__ Qg, const u16* __restrict__ Kg, const u16* __restrict__ Vg,
    u16* __restrict__ Ctx)
{
  __shared__ __align__(16) u16 Ks [224 * 72];    // K rows, stride 72 (filled by TDM)
  __shared__ __align__(16) u16 VTs[64 * 224];    // V transposed [d][key]
  __shared__ float gmx[NG * 64];
  __shared__ float gmn[NG * 64];
  __shared__ u32   selb[224];
  __shared__ __align__(16) u16 ATs[8 * 16 * 224]; // per-wave attn tiles

  const int tid  = threadIdx.x;
  const int lane = tid & 31, wid = tid >> 5;
  const int bh = blockIdx.x;
  const size_t base = (size_t)bh * NN * DD;
  const u16* Qp = Qg + base;
  const u16* Kp = Kg + base;
  const u16* Vp = Vg + base;

  // ---- K tile via Tensor Data Mover: 2D tensor 197x64 (u16), tile 224x64,
  //      OOB rows zero-filled, LDS pad 4 DWORDs per 32 DWORDs -> row stride 144B
  if (wid == 0) {
    u64 ga = (u64)(size_t)Kp;
    u32x4 g0;
    g0.x = 1u;                                         // count=1 valid descriptor
    g0.y = lds_off(Ks);                                // lds_addr
    g0.z = (u32)ga;                                    // global_addr[31:0]
    g0.w = (u32)((ga >> 32) & 0x01FFFFFFu) | (2u << 30); // addr[56:32] | type=2
    i32x8 g1 = {
      (int)((1u << 16) | (1u << 20) | (4u << 22) | (3u << 25)), // data_size=2B, pad_en, interval=32dw, amount=4dw
      (int)(64u  << 16),   // tensor_dim0 = 64  (low16 at bit48)
      (int)(197u << 16),   // tensor_dim0 hi=0 | tensor_dim1 = 197 (low16 at bit80)
      (int)(64u  << 16),   // tensor_dim1 hi=0 | tile_dim0 = 64
      (int)224,            // tile_dim1 = 224, tile_dim2 = 0
      (int)64,             // tensor_dim0_stride = 64 (low32)
      0, 0                 // stride hi / tensor_dim1_stride = 0
    };
    i32x4 z4 = {0, 0, 0, 0};
#if defined(__clang_major__) && (__clang_major__ >= 23)
    i32x8 z8 = {0, 0, 0, 0, 0, 0, 0, 0};
    __builtin_amdgcn_tensor_load_to_lds(g0, g1, z4, z4, z8, 0);
#else
    __builtin_amdgcn_tensor_load_to_lds(g0, g1, z4, z4, 0);
#endif
    __builtin_amdgcn_s_wait_tensorcnt(0);
  }

  // ---- load V transposed
  for (int e = tid; e < 64 * 224; e += 256) {
    int d = e / 224, key = e - d * 224;
    VTs[d * 224 + key] = (key < NN) ? Vp[key * 64 + d] : (u16)0;
  }
  __syncthreads();

  // ---- per-group key max/min (4 groups x 64 dims == 256 threads)
  {
    int g = tid >> 6, d = tid & 63;
    float mx = -__builtin_inff(), mn = __builtin_inff();
    for (int i = 0; i < GSZ; ++i) {
      float f = bf2f(Ks[(1 + g * GSZ + i) * 72 + d]);
      mx = fmaxf(mx, f); mn = fminf(mn, f);
    }
    gmx[g * 64 + d] = mx; gmn[g * 64 + d] = mn;
  }
  __syncthreads();

  // ---- per-query top-2-of-4 group selection
  if (tid < 224) {
    u32 sel = 0xFu;
    if (tid < NN) {
      float sc[NG] = {0.f, 0.f, 0.f, 0.f};
      for (int d = 0; d < 64; ++d) {
        float qv = bf2f(Qp[tid * 64 + d]);
        #pragma unroll
        for (int g = 0; g < NG; ++g)
          sc[g] += fmaxf(qv * gmx[g * 64 + d], qv * gmn[g * 64 + d]);
      }
      sel = 0u;
      #pragma unroll
      for (int g = 0; g < NG; ++g) {
        int cnt = 0;
        #pragma unroll
        for (int g2 = 0; g2 < NG; ++g2)
          cnt += (sc[g2] > sc[g]) || (sc[g2] == sc[g] && g2 < g);
        if (cnt < 2) sel |= (1u << g);
      }
    }
    selb[tid] = sel;
  }
  __syncthreads();

  u16* at = ATs + wid * (16 * 224);
  const int cl  = lane & 15;
  const int hi8 = (lane & 16) ? 8 : 0;

  for (int qt = wid; qt < 13; qt += 8) {          // 13 query tiles of 16
    const int qbase = qt * 16;

    // q A-fragments straight from global (row clamp, stores guarded later)
    FragU aq[2];
    {
      int qr = qbase + cl; if (qr > NN - 1) qr = NN - 1;
      #pragma unroll
      for (int s = 0; s < 2; ++s) {
        int k0 = s * 32 + ((lane & 16) ? 8 : 0);
        aq[s].q[0] = *(const uint4*)(Qp + qr * 64 + k0);
        aq[s].q[1] = *(const uint4*)(Qp + qr * 64 + k0 + 16);
      }
    }

    // ---- logits: 13 key tiles x 2 k-steps (D=64)
    v8f acc[13];
    #pragma unroll
    for (int j = 0; j < 13; ++j) {
      v8f c = (v8f){0,0,0,0,0,0,0,0};
      int key = j * 16 + cl;
      FragU b0, b1;
      { int k0 = 0  + ((lane & 16) ? 8 : 0);
        b0.q[0] = *(const uint4*)(Ks + key * 72 + k0);
        b0.q[1] = *(const uint4*)(Ks + key * 72 + k0 + 16); }
      { int k0 = 32 + ((lane & 16) ? 8 : 0);
        b1.q[0] = *(const uint4*)(Ks + key * 72 + k0);
        b1.q[1] = *(const uint4*)(Ks + key * 72 + k0 + 16); }
      c = wmma_bf16(aq[0].v, b0.v, c);
      c = wmma_bf16(aq[1].v, b1.v, c);
      acc[j] = c;
    }

    // ---- mask + scale (masked logits -> 0.0, padded cols -> -inf)
    u32 selp[8]; int rowp[8];
    #pragma unroll
    for (int p = 0; p < 8; ++p) { rowp[p] = qbase + p + hi8; selp[p] = selb[rowp[p]]; }
    #pragma unroll
    for (int j = 0; j < 13; ++j) {
      int col = j * 16 + cl;
      bool valid = col < NN;
      int g = (col - 1) / GSZ;   // unused when col==0
      #pragma unroll
      for (int p = 0; p < 8; ++p) {
        float l;
        if (!valid) l = -__builtin_inff();
        else {
          l = acc[j][p] * SCALEF;
          bool keep = (col == 0) || (rowp[p] == 0) || (((selp[p] >> g) & 1u) != 0u);
          if (!keep) l = 0.0f;
        }
        acc[j][p] = l;
      }
    }

    // ---- softmax over 197 columns (row spread across 16 lanes + 13 tiles)
    float inv[8];
    #pragma unroll
    for (int p = 0; p < 8; ++p) {
      float m = -__builtin_inff();
      #pragma unroll
      for (int j = 0; j < 13; ++j) m = fmaxf(m, acc[j][p]);
      #pragma unroll
      for (int off = 1; off <= 8; off <<= 1) m = fmaxf(m, __shfl_xor(m, off, 32));
      float s = 0.f;
      #pragma unroll
      for (int j = 0; j < 13; ++j) {
        float e = __expf(acc[j][p] - m);
        acc[j][p] = e; s += e;
      }
      #pragma unroll
      for (int off = 1; off <= 8; off <<= 1) s += __shfl_xor(s, off, 32);
      inv[p] = 1.0f / s;
    }

    // ---- normalized attn -> per-wave LDS tile (bf16), cols 208..223 zero
    #pragma unroll
    for (int j = 0; j < 14; ++j) {
      int col = j * 16 + cl;
      #pragma unroll
      for (int p = 0; p < 8; ++p) {
        float e = (j < 13) ? acc[j][p] * inv[p] : 0.0f;
        at[(p + hi8) * 224 + col] = f2bf(e);
      }
    }

    // ---- attn @ V : K-dim 224 = 7 steps of 32, 4 output d-chunks
    v8f oc[4];
    #pragma unroll
    for (int c = 0; c < 4; ++c) oc[c] = (v8f){0,0,0,0,0,0,0,0};
    #pragma unroll
    for (int ks = 0; ks < 7; ++ks) {
      FragU a;
      int k0 = ks * 32 + ((lane & 16) ? 8 : 0);
      a.q[0] = *(const uint4*)(at + cl * 224 + k0);
      a.q[1] = *(const uint4*)(at + cl * 224 + k0 + 16);
      #pragma unroll
      for (int c = 0; c < 4; ++c) {
        FragU bv;
        int dn  = c * 16 + cl;
        int k0b = ks * 32 + ((lane & 16) ? 8 : 0);
        bv.q[0] = *(const uint4*)(VTs + dn * 224 + k0b);
        bv.q[1] = *(const uint4*)(VTs + dn * 224 + k0b + 16);
        oc[c] = wmma_bf16(a.v, bv.v, oc[c]);
      }
    }

    // ---- store context bf16 [B,N, H*D]
    int h = bh % HH, b = bh / HH;
    #pragma unroll
    for (int c = 0; c < 4; ++c) {
      int d = c * 16 + cl;
      #pragma unroll
      for (int p = 0; p < 8; ++p) {
        int row = rowp[p];
        if (row < NN)
          Ctx[((size_t)(b * NN + row)) * CC + h * 64 + d] = f2bf(oc[c][p]);
      }
    }
  }
}

// ---------------------------------------------------------------- launcher
extern "C" void kernel_launch(void* const* d_in, const int* in_sizes, int n_in,
                              void* d_out, int out_size, void* d_ws, size_t ws_size,
                              hipStream_t stream) {
  const float* x     = (const float*)d_in[0];
  const float* wqkv  = (const float*)d_in[1];
  const float* bqkv  = (const float*)d_in[2];
  const float* wproj = (const float*)d_in[3];
  const float* bproj = (const float*)d_in[4];
  float* out = (float*)d_out;

  u16* ws = (u16*)d_ws;
  size_t o = 0;
  u16* xb     = ws + o; o += (size_t)MTOT * CC;        // 9.68M
  u16* wqkvb  = ws + o; o += (size_t)CC * 3 * CC;      // 1.77M
  u16* wprojb = ws + o; o += (size_t)CC * CC;          // 0.59M
  u16* qb     = ws + o; o += (size_t)BB * HH * NN * DD;
  u16* kb     = ws + o; o += (size_t)BB * HH * NN * DD;
  u16* vb     = ws + o; o += (size_t)BB * HH * NN * DD;
  u16* ctx    = ws + o; o += (size_t)MTOT * CC;
  (void)ws_size; (void)in_sizes; (void)n_in; (void)out_size;

  const int nx  = MTOT * CC;
  const int nwq = CC * 3 * CC;
  const int nwp = CC * CC;
  cvt_f32_bf16<<<(nx  + 255) / 256, 256, 0, stream>>>(x,     xb,     nx);
  cvt_f32_bf16<<<(nwq + 255) / 256, 256, 0, stream>>>(wqkv,  wqkvb,  nwq);
  cvt_f32_bf16<<<(nwp + 255) / 256, 256, 0, stream>>>(wproj, wprojb, nwp);

  // QKV GEMM: M=12608, K=768, N=2304
  gemm_bf16<<<dim3((3 * CC) / 64, (MTOT + 127) / 128), 256, 0, stream>>>(
      xb, wqkvb, bqkv, 3 * CC, 0, qb, kb, vb, (float*)nullptr);

  // fused masked attention per (b,h)
  attn_bf16<<<dim3(BB * HH), 256, 0, stream>>>(qb, kb, vb, ctx);

  // projection GEMM: M=12608, K=768, N=768
  gemm_bf16<<<dim3(CC / 64, (MTOT + 127) / 128), 256, 0, stream>>>(
      ctx, wprojb, bproj, CC, 1, (u16*)nullptr, (u16*)nullptr, (u16*)nullptr, out);
}
